// BinaryConv2d_80848464380614
// MI455X (gfx1250) — matrix operand order repaired and checked by reading, not compiled, not run on hardware
//
#include <hip/hip_runtime.h>
#include <hip/hip_bf16.h>

// BinaryConv2d on MI455X (gfx1250): implicit GEMM with v_wmma_f32_16x16x32_bf16.
// x (32,256,56,56) f32, w (256,256,3,3) -> sign(+-1). out (32,256,54,54) f32.
// Strategy: bf16 hi/lo split of x (exact products vs +-1 weights, f32 acc ->
// ~fp32 accuracy), NHWC repack (LDS tile transpose) so K (channels) is
// contiguous, weights pre-packed into WMMA A-fragment lane order. Whole GEMM
// working set (~104 MB) is L2-resident (192 MB).

typedef __bf16 v16bf __attribute__((ext_vector_type(16)));
typedef float  v8f   __attribute__((ext_vector_type(8)));

__device__ __forceinline__ unsigned int f32_to_bf16_rne(float f) {
    union { float f; unsigned int u; } v; v.f = f;
    unsigned int u = v.u;
    u += 0x7FFFu + ((u >> 16) & 1u);   // round-to-nearest-even (finite inputs)
    return u >> 16;
}
__device__ __forceinline__ float bf16_to_f32(unsigned int h) {
    union { unsigned int u; float f; } v; v.u = h << 16;
    return v.f;
}

// ---------------------------------------------------------------------------
// Pack sign(w) into WMMA A-fragment layout:
// apack flat index = ((mt*9 + tap)*8 + kt)*512 + lane*16 + el   (bf16 elements)
// A layout used here (contiguous k, matching B): lane<16 -> M=lane, K = {0..15};
//                        lane>=16 -> M=lane-16, K = {16..31}
// ---------------------------------------------------------------------------
__global__ __launch_bounds__(256) void bconv_pack_weights(
        const float* __restrict__ w, unsigned short* __restrict__ apack) {
    int idx = blockIdx.x * 256 + threadIdx.x;        // 0 .. 589823
    int el   = idx & 15;
    int lane = (idx >> 4) & 31;
    int kt   = (idx >> 9) & 7;
    int t    = (idx >> 12) % 9;
    int mt   = idx / 36864;                          // 16 m-tiles
    int mrow = mt * 16 + (lane & 15);
    int hi   = lane >> 4;
    int klocal = el + hi * 16;
    int c    = kt * 32 + klocal;                     // input channel
    int kh = t / 3, kw = t - 3 * (t / 3);
    float wv = w[((mrow * 256 + c) * 3 + kh) * 3 + kw];
    // jnp.sign semantics: sign(0) = 0
    unsigned short s = (wv > 0.f) ? (unsigned short)0x3F80u
                     : (wv < 0.f) ? (unsigned short)0xBF80u
                                  : (unsigned short)0x0000u;
    apack[idx] = s;
}

// ---------------------------------------------------------------------------
// NCHW f32 -> NHWC bf16 hi/lo split with LDS tile transpose.
// Tile: 64 channels x 64 pixels of the flat 56*56=3136 HW plane (49 tiles).
// Phase 1: coalesced float2 reads along q; pack (q,q+1) bf16 pair -> LDS uint.
// Phase 2: channel-contiguous uint4 (16B) NHWC stores, 128B per 4 threads.
// Grid: 32 n * 4 c-tiles * 49 q-tiles = 6272 blocks x 256 threads.
// ---------------------------------------------------------------------------
__global__ __launch_bounds__(256) void bconv_split_x(
        const float* __restrict__ x,
        unsigned short* __restrict__ xhi,
        unsigned short* __restrict__ xlo) {
    __shared__ unsigned int lds_hi[64 * 33];  // [c_local][q_pair], pad 33
    __shared__ unsigned int lds_lo[64 * 33];

    const int t  = threadIdx.x;
    const int b  = blockIdx.x;
    const int qt = b % 49;
    const int ct = (b / 49) & 3;
    const int n  = b / 196;
    const int q0 = qt * 64;
    const int c0 = ct * 64;

    // Phase 1: read 64c x 64q tile (float2 per thread-iter), split, stage.
    {
        const int tq = t & 31;                 // q-pair index 0..31
        const int cg = t >> 5;                 // 0..7
        const float* src = x + (n * 256 + c0) * 3136 + q0 + 2 * tq;
#pragma unroll
        for (int i = 0; i < 8; ++i) {
            const int c_local = cg * 8 + i;
            const float2 v = *(const float2*)(src + c_local * 3136);
            const unsigned int h0 = f32_to_bf16_rne(v.x);
            const unsigned int h1 = f32_to_bf16_rne(v.y);
            const unsigned int l0 = f32_to_bf16_rne(v.x - bf16_to_f32(h0));
            const unsigned int l1 = f32_to_bf16_rne(v.y - bf16_to_f32(h1));
            lds_hi[c_local * 33 + tq] = h0 | (h1 << 16);
            lds_lo[c_local * 33 + tq] = l0 | (l1 << 16);
        }
    }
    __syncthreads();

    // Phase 2: NHWC stores, channels contiguous. Thread writes 16 channels.
    {
        const int q_local = t >> 2;            // 0..63
        const int k4 = t & 3;                  // 16-channel slot within 64
        const int qh  = q_local >> 1;
        const int sel = q_local & 1;
        unsigned int vh[8], vl[8];
#pragma unroll
        for (int u = 0; u < 8; ++u) {
            const int c = k4 * 16 + 2 * u;
            unsigned int a = lds_hi[c * 33 + qh];
            unsigned int bb = lds_hi[(c + 1) * 33 + qh];
            unsigned int ah = sel ? (a >> 16) : (a & 0xFFFFu);
            unsigned int bh = sel ? (bb >> 16) : (bb & 0xFFFFu);
            vh[u] = ah | (bh << 16);
            a  = lds_lo[c * 33 + qh];
            bb = lds_lo[(c + 1) * 33 + qh];
            ah = sel ? (a >> 16) : (a & 0xFFFFu);
            bh = sel ? (bb >> 16) : (bb & 0xFFFFu);
            vl[u] = ah | (bh << 16);
        }
        const int off = (n * 3136 + q0 + q_local) * 256 + c0 + k4 * 16; // ushorts
        *(uint4*)(xhi + off)     = make_uint4(vh[0], vh[1], vh[2], vh[3]);
        *(uint4*)(xhi + off + 8) = make_uint4(vh[4], vh[5], vh[6], vh[7]);
        *(uint4*)(xlo + off)     = make_uint4(vl[0], vl[1], vl[2], vl[3]);
        *(uint4*)(xlo + off + 8) = make_uint4(vl[4], vl[5], vl[6], vl[7]);
    }
}

// ---------------------------------------------------------------------------
// Implicit-GEMM conv. Grid: 1458 blocks x 256 threads (8 waves).
// Wave tile: 32 out-channels x 64 pixels (2 m-tiles x 4 n-tiles).
// All 8 waves of a block share the same 64 pixels (full 256 channels).
// K loop: 9 taps x 8 chunks of 32 channels; hi and lo both accumulate.
// ---------------------------------------------------------------------------
__global__ __launch_bounds__(256) void bconv_wmma(
        const unsigned short* __restrict__ apack,
        const unsigned short* __restrict__ xhi,
        const unsigned short* __restrict__ xlo,
        float* __restrict__ out) {
    const int lane = threadIdx.x & 31;
    const int wave = threadIdx.x >> 5;
    const int job  = blockIdx.x * 8 + wave;          // 0 .. 11663
    const int mp   = job & 7;                        // m-pair: channels mp*32..+31
    const int nq   = job >> 3;                       // pixel group: nq*64..+63
    const int col  = lane & 15;
    const int half = lane >> 4;

    // Per-lane pixel addressing (each lane = one output pixel column per j)
    int base_el[4], outn[4], outrem[4];
#pragma unroll
    for (int j = 0; j < 4; ++j) {
        int p   = nq * 64 + j * 16 + col;            // flat pixel, < 93312
        int n   = p / 2916;                          // 54*54
        int rem = p - n * 2916;
        int oh  = rem / 54;
        int ow  = rem - oh * 54;
        base_el[j] = ((n * 56 + oh) * 56 + ow) * 256 + half * 16;  // NHWC bf16 idx
        outn[j] = n; outrem[j] = rem;
    }

    const unsigned short* a0p = apack + (mp * 2) * 36864 + lane * 16;

    v8f acc[2][4];
#pragma unroll
    for (int m = 0; m < 2; ++m)
#pragma unroll
        for (int j = 0; j < 4; ++j) acc[m][j] = {};

    for (int t = 0; t < 9; ++t) {                    // 3x3 taps
        const int tapoff = ((t / 3) * 56 + (t - 3 * (t / 3))) * 256;
        const unsigned short* at = a0p + t * 4096;
#pragma unroll
        for (int kt = 0; kt < 8; ++kt) {             // 256 channels / 32
            const int ko = tapoff + kt * 32;
            v16bf a0 = *(const v16bf*)(at + kt * 512);
            v16bf a1 = *(const v16bf*)(at + 36864 + kt * 512);
#pragma unroll
            for (int j = 0; j < 4; ++j) {
                v16bf bh = *(const v16bf*)(xhi + base_el[j] + ko);
                v16bf bl = *(const v16bf*)(xlo + base_el[j] + ko);
                // hi then lo, interleaved across m to maximize distance
                // between same-accumulator WMMAs (D->C chain slack).
                acc[0][j] = __builtin_amdgcn_wmma_f32_16x16x32_bf16(
                    false, a0, false, bh, (short)0, acc[0][j], false, false);
                acc[1][j] = __builtin_amdgcn_wmma_f32_16x16x32_bf16(
                    false, a1, false, bh, (short)0, acc[1][j], false, false);
                acc[0][j] = __builtin_amdgcn_wmma_f32_16x16x32_bf16(
                    false, a0, false, bl, (short)0, acc[0][j], false, false);
                acc[1][j] = __builtin_amdgcn_wmma_f32_16x16x32_bf16(
                    false, a1, false, bl, (short)0, acc[1][j], false, false);
            }
        }
    }

    // D layout: VGPR r holds M = r + 8*half, N = col. Output NCHW f32.
#pragma unroll
    for (int m = 0; m < 2; ++m) {
        const int ocb = (mp * 2 + m) * 16 + half * 8;
#pragma unroll
        for (int j = 0; j < 4; ++j) {
            const int base = (outn[j] * 256 + ocb) * 2916 + outrem[j];
#pragma unroll
            for (int r = 0; r < 8; ++r) {
                out[base + r * 2916] = acc[m][j][r];
            }
        }
    }
}

extern "C" void kernel_launch(void* const* d_in, const int* in_sizes, int n_in,
                              void* d_out, int out_size, void* d_ws, size_t ws_size,
                              hipStream_t stream) {
    const float* x = (const float*)d_in[0];   // (32,256,56,56)
    const float* w = (const float*)d_in[1];   // (256,256,3,3)
    float* out = (float*)d_out;               // (32,256,54,54)

    // Workspace layout: [0, 1.18MB) apack | [2MB, +51.4MB) xhi | then xlo
    unsigned short* apack = (unsigned short*)d_ws;
    unsigned short* xhi   = (unsigned short*)((char*)d_ws + (size_t)(2u << 20));
    unsigned short* xlo   = xhi + 25690112;   // 32*256*56*56

    bconv_pack_weights<<<2304, 256, 0, stream>>>(w, apack);     // 589824 elems
    bconv_split_x   <<<6272, 256, 0, stream>>>(x, xhi, xlo);    // 64c x 64q tiles
    bconv_wmma      <<<1458, 256, 0, stream>>>(apack, xhi, xlo, out);
}
